// CorrelationModule_15599321219140
// MI455X (gfx1250) — matrix operand order repaired and checked by reading, not compiled, not run on hardware
//
#include <hip/hip_runtime.h>
#include <hip/hip_bf16.h>

typedef __attribute__((ext_vector_type(16))) _Float16 v16h;
typedef __attribute__((ext_vector_type(8)))  _Float16 v8h;
typedef __attribute__((ext_vector_type(8)))  float    v8f;

#define NIMG 98   // b * K * K = 2 * 49
#define RAD  3

#define WMMA_F16(A, B, C) \
  __builtin_amdgcn_wmma_f32_16x16x32_f16(false, (A), false, (B), (short)0, (C), false, false)

// All f16 activation buffers use a halo-padded channels-last layout:
//   storage [N][(H+2)][(W+2)][C], logical pixel (y,x) at storage (y+1, x+1).
// The 1-pixel ring (rows -1,H and cols -1,W) is zeroed so conv loads need no
// bounds checks: y-taps always land in [-1,H]; x-taps of *valid* columns land
// in [-1,W]; junk columns of partial tiles are clamped to the zero column W
// (their results are never stored, and WMMA columns are independent).

// ---------------------------------------------------------------------------
// zero the halo ring of a padded activation buffer (one thread per halo pixel)
// ---------------------------------------------------------------------------
__global__ void zero_halo_k(_Float16* __restrict__ buf, int N, int H, int W, int C) {
  int SW = W + 2;
  int perimg = 2 * SW + 2 * H;
  int idx = blockIdx.x * blockDim.x + threadIdx.x;
  int total = N * perimg;
  if (idx >= total) return;
  int p = idx % perimg; int n = idx / perimg;
  int row, col;
  if (p < SW)          { row = 0;     col = p; }
  else if (p < 2 * SW) { row = H + 1; col = p - SW; }
  else { int q = p - 2 * SW; row = 1 + (q >> 1); col = (q & 1) ? (W + 1) : 0; }
  _Float16* d = buf + (((size_t)n * (H + 2) + row) * SW + col) * C;
  v8h z = {};
  for (int c = 0; c < C; c += 8) *(v8h*)(d + c) = z;
}

// ---------------------------------------------------------------------------
// coords avg-pool (exact: mean over 2^lvl x 2^lvl block == iterated 2x2 mean)
// cur layout: (b, 2, H2, W2) flat f32
// ---------------------------------------------------------------------------
__global__ void pool_coords_k(const float* __restrict__ coords, float* __restrict__ cur,
                              int H2, int W2, int s, int H, int W) {
  int idx = blockIdx.x * blockDim.x + threadIdx.x;
  int total = 2 * 2 * H2 * W2;
  if (idx >= total) return;
  int x2 = idx % W2; int t = idx / W2;
  int y2 = t % H2; t /= H2;
  int ch = t & 1; int b = t >> 1;
  const float* src = coords + ((size_t)(b * 2 + ch) * H) * W;
  float acc = 0.f;
  for (int dy = 0; dy < s; ++dy)
    for (int dx = 0; dx < s; ++dx)
      acc += src[(size_t)(y2 * s + dy) * W + (x2 * s + dx)];
  cur[idx] = acc / (float)(s * s);
}

// ---------------------------------------------------------------------------
// bilinear sample f2 at 49 displacements + concat f1 -> corr volume (padded)
// corr logical layout: [n=b*49][H][W][64] f16; ch 0..31 = f1, 32..63 = f2 sampled
// Reproduces the reference's raw-reshape coords quirk: px,py read flat as
// cur[b][(y*W+x)*2 + {0,1}] over the (2,H,W) block.
// ---------------------------------------------------------------------------
__global__ void sample_corr_k(const float* __restrict__ f1, const float* __restrict__ f2,
                              const float* __restrict__ cur, _Float16* __restrict__ corr,
                              int H, int W, int level) {
  int idx = blockIdx.x * blockDim.x + threadIdx.x;
  int total = NIMG * H * W;
  if (idx >= total) return;
  int x = idx % W; int t = idx / W;
  int y = t % H; int n = t / H;
  int a1 = n % 7, a0 = (n / 7) % 7, b = n / 49;
  const int HW = H * W;
  const float* curb = cur + (size_t)b * 2 * HW;
  float inv = 1.f / (float)(1 << level);
  float px = curb[(y * W + x) * 2 + 0] * inv + (float)(a0 - RAD);
  float py = curb[(y * W + x) * 2 + 1] * inv + (float)(a1 - RAD);
  float x0f = floorf(px), y0f = floorf(py);
  float wx = px - x0f, wy = py - y0f;
  int x0 = (int)x0f, y0 = (int)y0f;
  bool vx0 = (x0f >= 0.f) && (x0f <= (float)(W - 1));
  bool vx1 = (x0f + 1.f >= 0.f) && (x0f + 1.f <= (float)(W - 1));
  bool vy0 = (y0f >= 0.f) && (y0f <= (float)(H - 1));
  bool vy1 = (y0f + 1.f >= 0.f) && (y0f + 1.f <= (float)(H - 1));
  int x0c = min(max(x0, 0), W - 1);
  int x1c = min(max(x0 + 1, 0), W - 1);
  int y0c = min(max(y0, 0), H - 1);
  int y1c = min(max(y0 + 1, 0), H - 1);
  float w00 = (1.f - wy) * (1.f - wx), w01 = (1.f - wy) * wx;
  float w10 = wy * (1.f - wx),         w11 = wy * wx;
  const float* f1b = f1 + (size_t)b * 32 * HW + y * W + x;
  const float* f2b = f2 + (size_t)b * 32 * HW;
  _Float16* o = corr + (((size_t)n * (H + 2) + (y + 1)) * (W + 2) + (x + 1)) * 64;
  int o00 = y0c * W + x0c, o01 = y0c * W + x1c;
  int o10 = y1c * W + x0c, o11 = y1c * W + x1c;
  __align__(32) _Float16 buf[64];
#pragma unroll
  for (int c = 0; c < 32; ++c) {
    const float* fc = f2b + (size_t)c * HW;
    float v00 = (vy0 && vx0) ? fc[o00] : 0.f;
    float v01 = (vy0 && vx1) ? fc[o01] : 0.f;
    float v10 = (vy1 && vx0) ? fc[o10] : 0.f;
    float v11 = (vy1 && vx1) ? fc[o11] : 0.f;
    buf[c]      = (_Float16)f1b[(size_t)c * HW];
    buf[32 + c] = (_Float16)(v00 * w00 + v01 * w01 + v10 * w10 + v11 * w11);
  }
#pragma unroll
  for (int i = 0; i < 8; ++i)
    ((v8h*)o)[i] = ((const v8h*)buf)[i];
}

// ---------------------------------------------------------------------------
// Pack 3x3 conv weights (OIHW f32) into per-lane WMMA A fragments (f16).
// Fragment layout: [coT][tap(9)][kc][lane(32)][e(16)] ; A is 16(M=cout)x32(K=cin)
// A layout (matches contiguous B loads): element e (g=lane>>4) -> K = 16g+e
// ---------------------------------------------------------------------------
__global__ void pack_conv3x3_w_k(const float* __restrict__ w, _Float16* __restrict__ wpk,
                                 int Cout, int Cin) {
  int idx = blockIdx.x * blockDim.x + threadIdx.x;
  int kch = Cin >> 5;
  int total = (Cout >> 4) * 9 * kch * 512;
  if (idx >= total) return;
  int e = idx & 15;
  int lane = (idx >> 4) & 31;
  int rest = idx >> 9;
  int kc = rest % kch; rest /= kch;
  int tap = rest % 9; int coT = rest / 9;
  int m = lane & 15, g = lane >> 4, j = e >> 1;
  int kk = 16 * g + e; (void)j;
  int ci = kc * 32 + kk, co = coT * 16 + m;
  int ky = tap / 3, kx = tap % 3;
  wpk[idx] = (_Float16)w[((size_t)(co * Cin + ci) * 3 + ky) * 3 + kx];
}

// ConvTranspose2d weights (Cin=64, Cout=32, 4, 4) -> A fragments with the
// built-in spatial flip: value = w[ci][co][3-ky][3-kx].
// Fragment layout: [coT(2)][tap(16: ky*4+kx)][kc(2)][lane(32)][e(16)]
__global__ void pack_deconv_w_k(const float* __restrict__ w, _Float16* __restrict__ wpk) {
  int idx = blockIdx.x * blockDim.x + threadIdx.x;
  int total = 2 * 16 * 2 * 512;
  if (idx >= total) return;
  int e = idx & 15;
  int lane = (idx >> 4) & 31;
  int rest = idx >> 9;
  int kc = rest & 1; rest >>= 1;
  int tap = rest & 15; int coT = rest >> 4;
  int ky = tap >> 2, kx = tap & 3;
  int m = lane & 15, g = lane >> 4, j = e >> 1;
  int kk = 16 * g + e; (void)j;
  int ci = kc * 32 + kk, co = coT * 16 + m;
  wpk[idx] = (_Float16)w[((size_t)(ci * 32 + co) * 4 + (3 - ky)) * 4 + (3 - kx)];
}

// ---------------------------------------------------------------------------
// Implicit-GEMM 3x3 conv, pad 1, via v_wmma_f32_16x16x32_f16, with 2x2
// register tiling: each wave computes 32 couts x 32 pixels (4 acc tiles).
// Per K-step: 2 A loads + 2 B loads -> 4 WMMAs, all UNCONDITIONAL thanks to
// the zero halo (y-taps always in range; x clamped into the zero column).
// ---------------------------------------------------------------------------
template <int CIN, int COUT, int STRIDE, bool RELU>
__global__ void __launch_bounds__(256)
conv3x3_wmma_k(const _Float16* __restrict__ in, const _Float16* __restrict__ wpk,
               _Float16* __restrict__ out, int N, int Hin, int Win, int Hout, int Wout) {
  constexpr int KCH = CIN / 32;
  constexpr int NCT2 = COUT / 32;          // pairs of 16-cout A tiles
  const int lane = threadIdx.x & 31;
  const int nXT = (Wout + 31) >> 5;        // 32-pixel tiles
  int wid = blockIdx.x * (blockDim.x >> 5) + (threadIdx.x >> 5);
  int total = N * Hout * nXT * NCT2;
  if (wid >= total) return;
  int cp = wid % NCT2; int t = wid / NCT2;
  int xt = t % nXT; t /= nXT;
  int oy = t % Hout; int n = t / Hout;
  const int m = lane & 15, g = lane >> 4;
  const int ox0 = (xt << 5) + m;
  const int ox1 = ox0 + 16;
  const int SWin = Win + 2;
  v8f a00 = {}, a10 = {}, a01 = {}, a11 = {};
  const _Float16* wb0 = wpk + (size_t)(2 * cp) * 9 * KCH * 512 + lane * 16;
  const _Float16* wb1 = wb0 + (size_t)9 * KCH * 512;
#pragma unroll
  for (int ky = 0; ky < 3; ++ky) {
    int iy = oy * STRIDE + ky - 1;                       // always in [-1, Hin]
    const _Float16* rowb =
        in + (((size_t)(n * (Hin + 2) + (iy + 1))) * SWin + 1) * CIN + (g << 4);
#pragma unroll
    for (int kx = 0; kx < 3; ++kx) {
      int ix0 = min(ox0 * STRIDE + kx - 1, Win);         // junk cols -> zero col
      int ix1 = min(ox1 * STRIDE + kx - 1, Win);
      const _Float16* p0 = rowb + (ptrdiff_t)ix0 * CIN;  // ix = -1 hits halo col
      const _Float16* p1 = rowb + (ptrdiff_t)ix1 * CIN;
      const _Float16* wf0 = wb0 + (size_t)(ky * 3 + kx) * KCH * 512;
      const _Float16* wf1 = wb1 + (size_t)(ky * 3 + kx) * KCH * 512;
#pragma unroll
      for (int kc = 0; kc < KCH; ++kc) {
        v16h A0 = *(const v16h*)(wf0 + kc * 512);
        v16h A1 = *(const v16h*)(wf1 + kc * 512);
        v16h B0 = *(const v16h*)(p0 + kc * 32);
        v16h B1 = *(const v16h*)(p1 + kc * 32);
        a00 = WMMA_F16(A0, B0, a00);
        a10 = WMMA_F16(A1, B0, a10);
        a01 = WMMA_F16(A0, B1, a01);
        a11 = WMMA_F16(A1, B1, a11);
      }
    }
  }
  const int coT0 = 2 * cp;
  auto store_tile = [&](v8f acc, int coT, int ox) {
    if (ox < Wout) {
      v8h h;
#pragma unroll
      for (int v = 0; v < 8; ++v) {
        float f = acc[v];
        if (RELU) f = f > 0.f ? f : 0.f;
        h[v] = (_Float16)f;
      }
      *(v8h*)(out + (((size_t)(n * (Hout + 2) + (oy + 1))) * (Wout + 2) + (ox + 1)) * COUT
              + (coT << 4) + (g << 3)) = h;
    }
  };
  store_tile(a00, coT0,     ox0);
  store_tile(a10, coT0 + 1, ox0);
  store_tile(a01, coT0,     ox1);
  store_tile(a11, coT0 + 1, ox1);
}

// ---------------------------------------------------------------------------
// ConvTranspose2d k=4 s=2 p=1, Cin=64 -> Cout=32, via WMMA (2x2 tiling:
// 32 couts x 32 same-parity output columns per wave), halo-padded in/out.
// out[co,oy,ox] = sum over ky==oy (mod 2), kx==ox (mod 2):
//   w[ci,co,3-ky,3-kx] * in[ci,(oy+ky-2)/2,(ox+kx-2)/2]
// ---------------------------------------------------------------------------
template <bool RELU>
__global__ void __launch_bounds__(256)
deconv4x4_wmma_k(const _Float16* __restrict__ in, const _Float16* __restrict__ wpk,
                 _Float16* __restrict__ out, int N, int Hin, int Win, int Hout, int Wout) {
  const int lane = threadIdx.x & 31;
  const int m = lane & 15, g = lane >> 4;
  const int halfW = Wout >> 1;
  const int nXT = (halfW + 31) >> 5;
  int wid = blockIdx.x * (blockDim.x >> 5) + (threadIdx.x >> 5);
  int total = N * Hout * 2 * nXT;
  if (wid >= total) return;
  int xt = wid % nXT; int t = wid / nXT;
  int pxp = t & 1; t >>= 1;
  int oy = t % Hout; int n = t / Hout;
  int oxh0 = (xt << 5) + m, oxh1 = oxh0 + 16;
  int ox0 = (oxh0 << 1) + pxp, ox1 = ox0 + 32;
  int pyp = oy & 1;
  const int SWin = Win + 2;
  v8f a00 = {}, a10 = {}, a01 = {}, a11 = {};
#pragma unroll
  for (int ay = 0; ay < 2; ++ay) {
    int ky = pyp + 2 * ay;
    int iy = (oy + ky - 2) >> 1;                 // exact: oy+ky even; in [-1,Hin]
    const _Float16* rowb =
        in + (((size_t)(n * (Hin + 2) + (iy + 1))) * SWin + 1) * 64 + (g << 4);
#pragma unroll
    for (int ax = 0; ax < 2; ++ax) {
      int kx = pxp + 2 * ax;
      int ix0 = min((ox0 + kx - 2) >> 1, Win);   // exact: ox+kx even
      int ix1 = min(((ox0 + kx - 2) >> 1) + 16, Win);
      const _Float16* p0 = rowb + (ptrdiff_t)ix0 * 64;
      const _Float16* p1 = rowb + (ptrdiff_t)ix1 * 64;
      int tap = ky * 4 + kx;
#pragma unroll
      for (int kc = 0; kc < 2; ++kc) {
        v16h A0 = *(const v16h*)(wpk + ((size_t)tap * 2 + kc) * 512 + lane * 16);
        v16h A1 = *(const v16h*)(wpk + ((size_t)(16 + tap) * 2 + kc) * 512 + lane * 16);
        v16h B0 = *(const v16h*)(p0 + kc * 32);
        v16h B1 = *(const v16h*)(p1 + kc * 32);
        a00 = WMMA_F16(A0, B0, a00);
        a10 = WMMA_F16(A1, B0, a10);
        a01 = WMMA_F16(A0, B1, a01);
        a11 = WMMA_F16(A1, B1, a11);
      }
    }
  }
  auto store_tile = [&](v8f acc, int coT, int ox, int oxh) {
    if (oxh < halfW) {
      v8h h;
#pragma unroll
      for (int v = 0; v < 8; ++v) {
        float f = acc[v];
        if (RELU) f = f > 0.f ? f : 0.f;
        h[v] = (_Float16)f;
      }
      *(v8h*)(out + (((size_t)(n * (Hout + 2) + (oy + 1))) * (Wout + 2) + (ox + 1)) * 32
              + (coT << 4) + (g << 3)) = h;
    }
  };
  store_tile(a00, 0, ox0, oxh0);
  store_tile(a10, 1, ox0, oxh0);
  store_tile(a01, 0, ox1, oxh1);
  store_tile(a11, 1, ox1, oxh1);
}

// ---------------------------------------------------------------------------
// Final 3x3 conv 32 -> 1 (+ bias), f32 accumulate, halo-padded input (no
// bounds checks). cost layout: (b,49,H,W) f32 (unpadded).
// ---------------------------------------------------------------------------
__global__ void conv6_k(const _Float16* __restrict__ act, const float* __restrict__ w6,
                        const float* __restrict__ b6, float* __restrict__ cost,
                        int H, int W) {
  int idx = blockIdx.x * blockDim.x + threadIdx.x;
  int total = NIMG * H * W;
  if (idx >= total) return;
  int x = idx % W; int t = idx / W;
  int y = t % H; int n = t / H;
  const int SW = W + 2;
  float s = b6[0];
#pragma unroll
  for (int ky = 0; ky < 3; ++ky) {
#pragma unroll
    for (int kx = 0; kx < 3; ++kx) {
      const v16h* p = (const v16h*)(act
          + (((size_t)(n * (H + 2) + (y + ky))) * SW + (x + kx)) * 32);
      v16h c0 = p[0], c1 = p[1];
#pragma unroll
      for (int e = 0; e < 16; ++e) {
        s += (float)c0[e] * w6[(e * 3 + ky) * 3 + kx];
        s += (float)c1[e] * w6[((16 + e) * 3 + ky) * 3 + kx];
      }
    }
  }
  cost[idx] = s;
}

// DAP 49x49 1x1 mix + nearest upsample (repeat s x s) into output slice.
__global__ void dap_up_k(const float* __restrict__ cost, const float* __restrict__ dap,
                         float* __restrict__ outp, int H2, int W2, int s, int level) {
  int idx = blockIdx.x * blockDim.x + threadIdx.x;
  int total = 2 * 49 * H2 * W2;
  if (idx >= total) return;
  int x2 = idx % W2; int t = idx / W2;
  int y2 = t % H2; t /= H2;
  int o = t % 49; int b = t / 49;
  const float* cb = cost + ((size_t)b * 49) * H2 * W2 + (size_t)y2 * W2 + x2;
  float acc = 0.f;
  for (int c = 0; c < 49; ++c)
    acc += dap[o * 49 + c] * cb[(size_t)c * H2 * W2];
  float* ob = outp + (((size_t)(b * 196 + level * 49 + o)) * 64 + (size_t)y2 * s) * 96
            + (size_t)x2 * s;
  for (int dy = 0; dy < s; ++dy)
    for (int dx = 0; dx < s; ++dx)
      ob[(size_t)dy * 96 + dx] = acc;
}

// ---------------------------------------------------------------------------
// host orchestration
// ---------------------------------------------------------------------------
extern "C" void kernel_launch(void* const* d_in, const int* in_sizes, int n_in,
                              void* d_out, int out_size, void* d_ws, size_t ws_size,
                              hipStream_t stream) {
  // Identify inputs by flat element count (all sizes distinguishable; the three
  // 24576-element arrays disambiguate by order: fmap1_2, fmap2_2, coords).
  const float *f1[4] = {}, *f2[4] = {}, *coords = nullptr;
  const float *dapw[4] = {}, *b6p[4] = {}, *w1p[4] = {}, *w2p[4] = {};
  const float *w3p[4] = {}, *w4p[4] = {}, *w6p[4] = {}, *wdp[4] = {};
  int c393 = 0, c98 = 0, c24 = 0, c6 = 0, cd = 0, cb = 0;
  int c1 = 0, c2 = 0, c3 = 0, c4 = 0, c6w = 0, cw = 0;
  for (int i = 0; i < n_in; ++i) {
    const float* p = (const float*)d_in[i];
    switch (in_sizes[i]) {
      case 393216: { if (c393 == 0) f1[0] = p; else f2[0] = p; c393++; } break;
      case 98304:  { if (c98 == 0) f1[1] = p; else f2[1] = p; c98++; } break;
      case 24576:  { if (c24 == 0) f1[2] = p; else if (c24 == 1) f2[2] = p; else coords = p; c24++; } break;
      case 6144:   { if (c6 == 0) f1[3] = p; else f2[3] = p; c6++; } break;
      case 2401:   if (cd < 4) dapw[cd++] = p; break;
      case 1:      if (cb < 4) b6p[cb++] = p; break;
      case 55296:  if (c1 < 4) w1p[c1++] = p; break;
      case 110592: if (c2 < 4) w2p[c2++] = p; break;
      case 147456: if (c3 < 4) w3p[c3++] = p; break;
      case 73728:  if (c4 < 4) w4p[c4++] = p; break;
      case 288:    if (c6w < 4) w6p[c6w++] = p; break;
      case 32768:  if (cw < 4) wdp[cw++] = p; break;
      default: break;
    }
  }

  // Workspace layout (peak ~224 MB): two big ping-pong regions for activations.
  char* ws = (char*)d_ws;
  float*    cur  = (float*)(ws);                      // 128 KB slot
  _Float16* w1k  = (_Float16*)(ws + (128u << 10));    // 256 KB slot (needs 108 KB)
  _Float16* w2k  = (_Float16*)(ws + (384u << 10));    // 256 KB slot (216 KB)
  _Float16* w3k  = (_Float16*)(ws + (640u << 10));    // 384 KB slot (288 KB)
  _Float16* w4k  = (_Float16*)(ws + (1024u << 10));   // 256 KB slot (144 KB)
  _Float16* wdk  = (_Float16*)(ws + (1280u << 10));   // 128 KB slot (64 KB)
  float*    cost = (float*)(ws + (1536u << 10));      // 3 MB slot (needs 2.3 MB)
  char*     regA = ws + (8ull << 20);                 // 88 MB (corr / act2 / act4)
  char*     regB = ws + (96ull << 20);                // 128 MB (act1 / act3 / act5)

  auto cdiv = [](int a, int b) { return (a + b - 1) / b; };

  for (int lvl = 0; lvl < 4; ++lvl) {
    int H2 = 64 >> lvl, W2 = 96 >> lvl, s = 1 << lvl;
    int H4 = H2 >> 1, W4 = W2 >> 1;

    // pack weights for this level into WMMA fragment layouts
    { int tot = (96 / 16) * 9 * (64 / 32) * 512;
      pack_conv3x3_w_k<<<cdiv(tot, 256), 256, 0, stream>>>(w1p[lvl], w1k, 96, 64); }
    { int tot = (128 / 16) * 9 * (96 / 32) * 512;
      pack_conv3x3_w_k<<<cdiv(tot, 256), 256, 0, stream>>>(w2p[lvl], w2k, 128, 96); }
    { int tot = (128 / 16) * 9 * (128 / 32) * 512;
      pack_conv3x3_w_k<<<cdiv(tot, 256), 256, 0, stream>>>(w3p[lvl], w3k, 128, 128); }
    { int tot = (64 / 16) * 9 * (128 / 32) * 512;
      pack_conv3x3_w_k<<<cdiv(tot, 256), 256, 0, stream>>>(w4p[lvl], w4k, 64, 128); }
    { int tot = 2 * 16 * 2 * 512;
      pack_deconv_w_k<<<cdiv(tot, 256), 256, 0, stream>>>(wdp[lvl], wdk); }

    // pool coords, build corr volume (f16 channels-last, halo-padded)
    { int tot = 2 * 2 * H2 * W2;
      pool_coords_k<<<cdiv(tot, 256), 256, 0, stream>>>(coords, cur, H2, W2, s, 64, 96); }
    _Float16* corr = (_Float16*)regA;
    { int tot = NIMG * (2 * (W2 + 2) + 2 * H2);
      zero_halo_k<<<cdiv(tot, 256), 256, 0, stream>>>(corr, NIMG, H2, W2, 64); }
    { int tot = NIMG * H2 * W2;
      sample_corr_k<<<cdiv(tot, 256), 256, 0, stream>>>(f1[lvl], f2[lvl], cur, corr, H2, W2, lvl); }

    // matching net (all WMMA, 2x2 register tiling: 32 cout x 32 px per wave)
    _Float16* act1 = (_Float16*)regB;
    { int tot = NIMG * (2 * (W2 + 2) + 2 * H2);
      zero_halo_k<<<cdiv(tot, 256), 256, 0, stream>>>(act1, NIMG, H2, W2, 96); }
    { int waves = NIMG * H2 * cdiv(W2, 32) * (96 / 32);
      conv3x3_wmma_k<64, 96, 1, true><<<cdiv(waves, 8), 256, 0, stream>>>(
          corr, w1k, act1, NIMG, H2, W2, H2, W2); }
    _Float16* act2 = (_Float16*)regA;
    { int tot = NIMG * (2 * (W4 + 2) + 2 * H4);
      zero_halo_k<<<cdiv(tot, 256), 256, 0, stream>>>(act2, NIMG, H4, W4, 128); }
    { int waves = NIMG * H4 * cdiv(W4, 32) * (128 / 32);
      conv3x3_wmma_k<96, 128, 2, true><<<cdiv(waves, 8), 256, 0, stream>>>(
          act1, w2k, act2, NIMG, H2, W2, H4, W4); }
    _Float16* act3 = (_Float16*)regB;
    { int tot = NIMG * (2 * (W4 + 2) + 2 * H4);
      zero_halo_k<<<cdiv(tot, 256), 256, 0, stream>>>(act3, NIMG, H4, W4, 128); }
    { int waves = NIMG * H4 * cdiv(W4, 32) * (128 / 32);
      conv3x3_wmma_k<128, 128, 1, true><<<cdiv(waves, 8), 256, 0, stream>>>(
          act2, w3k, act3, NIMG, H4, W4, H4, W4); }
    _Float16* act4 = (_Float16*)regA;
    { int tot = NIMG * (2 * (W4 + 2) + 2 * H4);
      zero_halo_k<<<cdiv(tot, 256), 256, 0, stream>>>(act4, NIMG, H4, W4, 64); }
    { int waves = NIMG * H4 * cdiv(W4, 32) * (64 / 32);
      conv3x3_wmma_k<128, 64, 1, true><<<cdiv(waves, 8), 256, 0, stream>>>(
          act3, w4k, act4, NIMG, H4, W4, H4, W4); }
    _Float16* act5 = (_Float16*)regB;
    { int tot = NIMG * (2 * (W2 + 2) + 2 * H2);
      zero_halo_k<<<cdiv(tot, 256), 256, 0, stream>>>(act5, NIMG, H2, W2, 32); }
    { int waves = NIMG * H2 * 2 * cdiv(W2 / 2, 32);
      deconv4x4_wmma_k<true><<<cdiv(waves, 8), 256, 0, stream>>>(
          act4, wdk, act5, NIMG, H4, W4, H2, W2); }

    // conv6 + bias (halo-padded input, no guards), then DAP mix + upsample
    { int tot = NIMG * H2 * W2;
      conv6_k<<<cdiv(tot, 256), 256, 0, stream>>>(act5, w6p[lvl], b6p[lvl], cost, H2, W2); }
    { int tot = 2 * 49 * H2 * W2;
      dap_up_k<<<cdiv(tot, 256), 256, 0, stream>>>(cost, dapw[lvl], (float*)d_out, H2, W2, s, lvl); }
  }
  (void)ws_size; (void)out_size; (void)n_in;
}
